// WaveEngine_89721866813817
// MI455X (gfx1250) — compile-verified, matrix operand order checked by reading, not run on hardware
//
#include <hip/hip_runtime.h>
#include <hip/hip_bf16.h>
#include <math.h>

// Problem dims (fixed by the reference)
#define BB 2
#define SS 2048
#define DD 1024
#define FF 64
#define TT 28
#define VV 50257
#define RR (BB*SS)          // 4096 rows (B*S)
#define NG_LM 786           // ceil(V/64) N-groups for LM head (64 cols per wave)

typedef __attribute__((ext_vector_type(16))) __bf16 v16bf;
typedef __attribute__((ext_vector_type(8)))  float  v8f;

union Frag16 { v16bf v; uint4 u[2]; };

__device__ __forceinline__ unsigned short f2bf(float x) {
    unsigned int u = __float_as_uint(x);
    u += 0x7FFFu + ((u >> 16) & 1u);     // round-to-nearest-even
    return (unsigned short)(u >> 16);
}

#define WMMA_BF16(a, b, c) \
    __builtin_amdgcn_wmma_f32_16x16x32_bf16(false, (a), false, (b), (short)0, (c), false, false)

// Load a 16x32 bf16 fragment (A-row / B-col striping per CDNA5 ISA 7.12.2)
// from a row-major bf16 matrix with leading dim `ld` (multiple of 8).
__device__ __forceinline__ v16bf load_frag(const unsigned short* __restrict__ base,
                                           int row0, int ld, int k0) {
    int lane = threadIdx.x & 31;
    int r    = lane & 15;
    int half = lane >> 4;
    const unsigned short* p = base + (size_t)(row0 + r) * ld + k0 + half * 8;
    Frag16 f;
    f.u[0] = *(const uint4*)(p);
    f.u[1] = *(const uint4*)(p + 16);
    return f.v;
}

__device__ __forceinline__ v16bf load_frag_guard(const unsigned short* __restrict__ base,
                                                 int row0, int nrows, int ld, int k0) {
    int lane = threadIdx.x & 31;
    int r    = lane & 15;
    int half = lane >> 4;
    int row  = row0 + r;
    Frag16 f;
    if (row < nrows) {
        const unsigned short* p = base + (size_t)row * ld + k0 + half * 8;
        f.u[0] = *(const uint4*)(p);
        f.u[1] = *(const uint4*)(p + 16);
    } else {
        f.u[0] = make_uint4(0u, 0u, 0u, 0u);
        f.u[1] = make_uint4(0u, 0u, 0u, 0u);
    }
    return f.v;
}

// ---------------- LM head (defined FIRST so its loop shows in the disasm) ----

// 32x64 tile body; GUARD=true only for the single N-group straddling V.
template <bool GUARD>
__device__ __forceinline__ void lm_tile(const unsigned short* __restrict__ A,
                                        const unsigned short* __restrict__ Bw,
                                        float* __restrict__ out, int m0, int n0) {
    v8f acc[8] = {};

    // set X <- k = 0
    v16bf xa0 = load_frag(A, m0,      DD, 0);
    v16bf xa1 = load_frag(A, m0 + 16, DD, 0);
    v16bf xb0, xb1, xb2, xb3, ya0, ya1, yb0, yb1, yb2, yb3;
    if (GUARD) {
        xb0 = load_frag_guard(Bw, n0,      VV, DD, 0);
        xb1 = load_frag_guard(Bw, n0 + 16, VV, DD, 0);
        xb2 = load_frag_guard(Bw, n0 + 32, VV, DD, 0);
        xb3 = load_frag_guard(Bw, n0 + 48, VV, DD, 0);
    } else {
        xb0 = load_frag(Bw, n0,      DD, 0);
        xb1 = load_frag(Bw, n0 + 16, DD, 0);
        xb2 = load_frag(Bw, n0 + 32, DD, 0);
        xb3 = load_frag(Bw, n0 + 48, DD, 0);
    }

#pragma unroll 1
    for (int k0 = 0; k0 < DD - 64; k0 += 64) {         // 15 iterations (k-steps 0..928)
        // phase 1: prefetch Y <- k0+32, compute with X (k0)
        ya0 = load_frag(A, m0,      DD, k0 + 32);
        ya1 = load_frag(A, m0 + 16, DD, k0 + 32);
        if (GUARD) {
            yb0 = load_frag_guard(Bw, n0,      VV, DD, k0 + 32);
            yb1 = load_frag_guard(Bw, n0 + 16, VV, DD, k0 + 32);
            yb2 = load_frag_guard(Bw, n0 + 32, VV, DD, k0 + 32);
            yb3 = load_frag_guard(Bw, n0 + 48, VV, DD, k0 + 32);
        } else {
            yb0 = load_frag(Bw, n0,      DD, k0 + 32);
            yb1 = load_frag(Bw, n0 + 16, DD, k0 + 32);
            yb2 = load_frag(Bw, n0 + 32, DD, k0 + 32);
            yb3 = load_frag(Bw, n0 + 48, DD, k0 + 32);
        }
        acc[0] = WMMA_BF16(xa0, xb0, acc[0]);
        acc[1] = WMMA_BF16(xa1, xb0, acc[1]);
        acc[2] = WMMA_BF16(xa0, xb1, acc[2]);
        acc[3] = WMMA_BF16(xa1, xb1, acc[3]);
        acc[4] = WMMA_BF16(xa0, xb2, acc[4]);
        acc[5] = WMMA_BF16(xa1, xb2, acc[5]);
        acc[6] = WMMA_BF16(xa0, xb3, acc[6]);
        acc[7] = WMMA_BF16(xa1, xb3, acc[7]);
        // phase 2: prefetch X <- k0+64, compute with Y (k0+32)
        xa0 = load_frag(A, m0,      DD, k0 + 64);
        xa1 = load_frag(A, m0 + 16, DD, k0 + 64);
        if (GUARD) {
            xb0 = load_frag_guard(Bw, n0,      VV, DD, k0 + 64);
            xb1 = load_frag_guard(Bw, n0 + 16, VV, DD, k0 + 64);
            xb2 = load_frag_guard(Bw, n0 + 32, VV, DD, k0 + 64);
            xb3 = load_frag_guard(Bw, n0 + 48, VV, DD, k0 + 64);
        } else {
            xb0 = load_frag(Bw, n0,      DD, k0 + 64);
            xb1 = load_frag(Bw, n0 + 16, DD, k0 + 64);
            xb2 = load_frag(Bw, n0 + 32, DD, k0 + 64);
            xb3 = load_frag(Bw, n0 + 48, DD, k0 + 64);
        }
        acc[0] = WMMA_BF16(ya0, yb0, acc[0]);
        acc[1] = WMMA_BF16(ya1, yb0, acc[1]);
        acc[2] = WMMA_BF16(ya0, yb1, acc[2]);
        acc[3] = WMMA_BF16(ya1, yb1, acc[3]);
        acc[4] = WMMA_BF16(ya0, yb2, acc[4]);
        acc[5] = WMMA_BF16(ya1, yb2, acc[5]);
        acc[6] = WMMA_BF16(ya0, yb3, acc[6]);
        acc[7] = WMMA_BF16(ya1, yb3, acc[7]);
    }
    // peeled tail: k-steps DD-64 (in X) and DD-32 (prefetch into Y)
    ya0 = load_frag(A, m0,      DD, DD - 32);
    ya1 = load_frag(A, m0 + 16, DD, DD - 32);
    if (GUARD) {
        yb0 = load_frag_guard(Bw, n0,      VV, DD, DD - 32);
        yb1 = load_frag_guard(Bw, n0 + 16, VV, DD, DD - 32);
        yb2 = load_frag_guard(Bw, n0 + 32, VV, DD, DD - 32);
        yb3 = load_frag_guard(Bw, n0 + 48, VV, DD, DD - 32);
    } else {
        yb0 = load_frag(Bw, n0,      DD, DD - 32);
        yb1 = load_frag(Bw, n0 + 16, DD, DD - 32);
        yb2 = load_frag(Bw, n0 + 32, DD, DD - 32);
        yb3 = load_frag(Bw, n0 + 48, DD, DD - 32);
    }
    acc[0] = WMMA_BF16(xa0, xb0, acc[0]);
    acc[1] = WMMA_BF16(xa1, xb0, acc[1]);
    acc[2] = WMMA_BF16(xa0, xb1, acc[2]);
    acc[3] = WMMA_BF16(xa1, xb1, acc[3]);
    acc[4] = WMMA_BF16(xa0, xb2, acc[4]);
    acc[5] = WMMA_BF16(xa1, xb2, acc[5]);
    acc[6] = WMMA_BF16(xa0, xb3, acc[6]);
    acc[7] = WMMA_BF16(xa1, xb3, acc[7]);
    acc[0] = WMMA_BF16(ya0, yb0, acc[0]);
    acc[1] = WMMA_BF16(ya1, yb0, acc[1]);
    acc[2] = WMMA_BF16(ya0, yb1, acc[2]);
    acc[3] = WMMA_BF16(ya1, yb1, acc[3]);
    acc[4] = WMMA_BF16(ya0, yb2, acc[4]);
    acc[5] = WMMA_BF16(ya1, yb2, acc[5]);
    acc[6] = WMMA_BF16(ya0, yb3, acc[6]);
    acc[7] = WMMA_BF16(ya1, yb3, acc[7]);

    int lane = threadIdx.x & 31;
    int cbase = lane & 15;
    int rbase = (lane >> 4) * 8;
#pragma unroll
    for (int j = 0; j < 4; ++j) {
        int col = n0 + j * 16 + cbase;
        if (!GUARD || col < VV) {
#pragma unroll
            for (int i = 0; i < 8; ++i) {
                out[(size_t)(m0 + rbase + i) * VV + col]      = acc[2*j][i];
                out[(size_t)(m0 + 16 + rbase + i) * VV + col] = acc[2*j+1][i];
            }
        }
    }
}

// out[R,V] = (a bf16 [R,1024]) x lm_head^T ; B frag == row-major lm_head_W[v,d].
// One wave computes a 32x64 tile: 2 A-frags x 4 B-frags -> 8 WMMAs per k-step
// (1.5 b128 loads per WMMA, ~22 FLOP/B from L2; lm_head bf16 weights are L2-resident).
// Only the single N-group straddling V takes the guarded path.
__global__ __launch_bounds__(256) void k_lm_head(const unsigned short* __restrict__ A,
                                                 const unsigned short* __restrict__ Bw,
                                                 float* __restrict__ out) {
    int wave = (blockIdx.x * 256 + threadIdx.x) >> 5;  // 100608 waves = 128 Mg x 786 Ng
    int ng = wave % NG_LM;
    int mg = wave / NG_LM;
    if (mg >= RR / 32) return;                         // wave-uniform (exact grid anyway)
    int m0 = mg * 32;
    int n0 = ng * 64;
    if (n0 + 64 <= VV) lm_tile<false>(A, Bw, out, m0, n0);  // 785/786 groups: no guards
    else               lm_tile<true >(A, Bw, out, m0, n0);  // edge group only
}

// ---------------- elementwise / setup kernels ----------------

__global__ __launch_bounds__(256) void k_f32_to_bf16(const float* __restrict__ in,
                                                     unsigned short* __restrict__ out, int n) {
    int i = blockIdx.x * 256 + threadIdx.x;
    if (i < n) out[i] = f2bf(in[i]);
}

__global__ __launch_bounds__(256) void k_embed(const int* __restrict__ tokens,
                                               const float* __restrict__ embW,
                                               float* __restrict__ x) {
    int r = blockIdx.x;                       // one block per row
    int tok = tokens[r];
    const float4* src = (const float4*)(embW + (size_t)tok * DD);
    float4* dst = (float4*)(x + (size_t)r * DD);
    for (int i = threadIdx.x; i < DD / 4; i += 256) dst[i] = src[i];
}

__global__ __launch_bounds__(256) void k_rms(const float* __restrict__ x,
                                             float* __restrict__ rowscale) {
    __shared__ float red[256];
    int r = blockIdx.x;
    const float* row = x + (size_t)r * DD;
    float s = 0.f;
    for (int d = threadIdx.x; d < DD; d += 256) { float v = row[d]; s += v * v; }
    red[threadIdx.x] = s;
    __syncthreads();
    for (int o = 128; o > 0; o >>= 1) {
        if (threadIdx.x < o) red[threadIdx.x] += red[threadIdx.x + o];
        __syncthreads();
    }
    if (threadIdx.x == 0)
        rowscale[r] = rsqrtf(red[0] * (1.0f / DD) + 1.1920929e-7f);  // torch fp32 eps
}

__global__ __launch_bounds__(256) void k_scale_bf16(const float* __restrict__ x,
                                                    const float* __restrict__ rowscale,
                                                    const float* __restrict__ w,
                                                    unsigned short* __restrict__ a) {
    int i = blockIdx.x * 256 + threadIdx.x;   // over R*D
    if (i < RR * DD) {
        int r = i >> 10, d = i & 1023;
        a[i] = f2bf(x[i] * rowscale[r] * w[d]);
    }
}

// ---------------- step-loop WMMA GEMM kernels ----------------

// wc[R,128] = A(bf16 [R,1024]) x to_wave^T ; B fragment == row-major to_wave_W[f2,d]
// One wave computes 16x64 (4 N-tiles); double-buffered.
__global__ __launch_bounds__(256) void k_gemm_towave(const unsigned short* __restrict__ A,
                                                     const unsigned short* __restrict__ Bw,
                                                     float* __restrict__ wc) {
    int wave = (blockIdx.x * 256 + threadIdx.x) >> 5;   // 512 waves = 256 Mt x 2 Ng
    int ng = wave & 1;
    int mt = wave >> 1;
    int m0 = mt * 16, n0 = ng * 64;
    v8f acc[4] = {};

    v16bf a  = load_frag(A, m0, DD, 0);
    v16bf b0 = load_frag(Bw, n0,      DD, 0);
    v16bf b1 = load_frag(Bw, n0 + 16, DD, 0);
    v16bf b2 = load_frag(Bw, n0 + 32, DD, 0);
    v16bf b3 = load_frag(Bw, n0 + 48, DD, 0);

#pragma unroll 1
    for (int k0 = 0; k0 < DD - 32; k0 += 32) {
        int kn = k0 + 32;
        v16bf na  = load_frag(A, m0, DD, kn);
        v16bf nb0 = load_frag(Bw, n0,      DD, kn);
        v16bf nb1 = load_frag(Bw, n0 + 16, DD, kn);
        v16bf nb2 = load_frag(Bw, n0 + 32, DD, kn);
        v16bf nb3 = load_frag(Bw, n0 + 48, DD, kn);
        acc[0] = WMMA_BF16(a, b0, acc[0]);
        acc[1] = WMMA_BF16(a, b1, acc[1]);
        acc[2] = WMMA_BF16(a, b2, acc[2]);
        acc[3] = WMMA_BF16(a, b3, acc[3]);
        a = na; b0 = nb0; b1 = nb1; b2 = nb2; b3 = nb3;
    }
    acc[0] = WMMA_BF16(a, b0, acc[0]);
    acc[1] = WMMA_BF16(a, b1, acc[1]);
    acc[2] = WMMA_BF16(a, b2, acc[2]);
    acc[3] = WMMA_BF16(a, b3, acc[3]);

    int lane = threadIdx.x & 31;
    int cbase = lane & 15;
    int rbase = m0 + (lane >> 4) * 8;
#pragma unroll
    for (int j = 0; j < 4; ++j) {
        int col = n0 + j * 16 + cbase;
#pragma unroll
        for (int i = 0; i < 8; ++i)
            wc[(size_t)(rbase + i) * (2 * FF) + col] = acc[j][i];
    }
}

// waves = complex(wc[:, :F], wc[:, F:]) * exp(i*s*pos_freq) ; transpose to [(b,f), s]
__global__ __launch_bounds__(256) void k_make_waves(const float* __restrict__ wc,
                                                    const float* __restrict__ pos_freq,
                                                    float* __restrict__ wre,
                                                    float* __restrict__ wim) {
    int i = blockIdx.x * 256 + threadIdx.x;   // R*F
    if (i >= RR * FF) return;
    int r = i >> 6, f = i & 63;
    int s = r & (SS - 1), b = r >> 11;
    float wr = wc[(size_t)r * 128 + f];
    float wi = wc[(size_t)r * 128 + FF + f];
    float sn, cs;
    __sincosf((float)s * pos_freq[f], &sn, &cs);
    size_t o = (size_t)(b * FF + f) * SS + s;
    wre[o] = wr * cs - wi * sn;
    wim[o] = wr * sn + wi * cs;
}

// 2048-point radix-2 FFT in LDS, one column (b,f) per block, 1024 threads.
// mode 0: forward (exp(-i..)), multiply by mix[f], write [(b,f),s] complex.
// mode 1: inverse (exp(+i..))/N, write real part to out_re as [R,F] (row b*S+s, col f).
__global__ __launch_bounds__(1024) void k_fft2048(const float* __restrict__ in_re,
                                                  const float* __restrict__ in_im,
                                                  float* __restrict__ out_re,
                                                  float* __restrict__ out_im,
                                                  const float* __restrict__ speed,
                                                  const float* __restrict__ dampen,
                                                  const float* __restrict__ phase,
                                                  int mode) {
    __shared__ float sre[SS];
    __shared__ float sim[SS];
    int col = blockIdx.x;             // b*F + f
    int f = col & (FF - 1);
    int b = col >> 6;
    int tid = threadIdx.x;
    const float* ir = in_re + (size_t)col * SS;
    const float* ii = in_im + (size_t)col * SS;
    for (int i = tid; i < SS; i += 1024) {
        int j = (int)(__brev((unsigned)i) >> 21);   // 11-bit bit reversal
        sre[j] = ir[i];
        sim[j] = ii[i];
    }
    __syncthreads();
    const float sgn = (mode == 0) ? -1.0f : 1.0f;
    for (int len = 2; len <= SS; len <<= 1) {
        int half = len >> 1;
        float ang0 = sgn * 6.28318530717958647692f / (float)len;
        int grp = tid / half;
        int pos = tid - grp * half;
        int i0 = grp * len + pos, i1 = i0 + half;
        float wsn, wcs;
        __sincosf(ang0 * (float)pos, &wsn, &wcs);
        float xr = sre[i1], xi = sim[i1];
        float tr = xr * wcs - xi * wsn;
        float ti = xr * wsn + xi * wcs;
        float ur = sre[i0], ui = sim[i0];
        sre[i0] = ur + tr; sim[i0] = ui + ti;
        sre[i1] = ur - tr; sim[i1] = ui - ti;
        __syncthreads();
    }
    if (mode == 0) {
        // mix = exp(i*(speed*pi + phase)) * exp(-softplus(dampen))
        float dm = dampen[f];
        float theta = speed[f] * 3.14159265358979323846f + phase[f];
        float sp = (dm > 20.0f) ? dm : log1pf(__expf(dm));
        float amp = __expf(-sp);
        float mr = amp * __cosf(theta);
        float mi = amp * __sinf(theta);
        for (int i = tid; i < SS; i += 1024) {
            float xr = sre[i], xi = sim[i];
            out_re[(size_t)col * SS + i] = xr * mr - xi * mi;
            out_im[(size_t)col * SS + i] = xr * mi + xi * mr;
        }
    } else {
        const float inv = 1.0f / (float)SS;
        for (int i = tid; i < SS; i += 1024)
            out_re[(size_t)(b * SS + i) * FF + f] = sre[i] * inv;   // [R,F] real
    }
}

// wg[(b,g),s] = sum_f coupling[f,g] * wf[(b,f),s]   (coupling is real)
__global__ __launch_bounds__(256) void k_coupling(const float* __restrict__ wf_re,
                                                  const float* __restrict__ wf_im,
                                                  const float* __restrict__ coupling,
                                                  float* __restrict__ wg_re,
                                                  float* __restrict__ wg_im) {
    int i = blockIdx.x * 256 + threadIdx.x;   // B*F*S
    if (i >= BB * FF * SS) return;
    int s = i & (SS - 1);
    int g = (i >> 11) & (FF - 1);
    int b = i >> 17;
    float ar = 0.f, ai = 0.f;
#pragma unroll 4
    for (int f = 0; f < FF; ++f) {
        float c = coupling[f * FF + g];
        size_t o = (size_t)(b * FF + f) * SS + s;
        ar += c * wf_re[o];
        ai += c * wf_im[o];
    }
    size_t o = (size_t)(b * FF + g) * SS + s;
    wg_re[o] = ar;
    wg_im[o] = ai;
}

// x[R,D] += 0.1*ss[t] * (wro bf16 [R,64]) x (from_wave^T)
// One wave computes 16x64 (4 N-tiles). K=64 -> only 2 k-steps: issue ALL 10
// fragment loads up front, then run the 8 WMMAs (no load/WMMA serialization).
__global__ __launch_bounds__(256) void k_delta_update(const unsigned short* __restrict__ A,
                                                      const unsigned short* __restrict__ Bw,
                                                      const float* __restrict__ step_scale, int t,
                                                      float* __restrict__ x) {
    int wave = (blockIdx.x * 256 + threadIdx.x) >> 5;  // 4096 waves = 256 Mt x 16 Ng
    int ng = wave & 15;
    int mt = wave >> 4;
    int m0 = mt * 16, n0 = ng * 64;

    v16bf a0  = load_frag(A, m0, FF, 0);
    v16bf b00 = load_frag(Bw, n0,      FF, 0);
    v16bf b01 = load_frag(Bw, n0 + 16, FF, 0);
    v16bf b02 = load_frag(Bw, n0 + 32, FF, 0);
    v16bf b03 = load_frag(Bw, n0 + 48, FF, 0);
    v16bf a1  = load_frag(A, m0, FF, 32);
    v16bf b10 = load_frag(Bw, n0,      FF, 32);
    v16bf b11 = load_frag(Bw, n0 + 16, FF, 32);
    v16bf b12 = load_frag(Bw, n0 + 32, FF, 32);
    v16bf b13 = load_frag(Bw, n0 + 48, FF, 32);

    v8f acc[4] = {};
    acc[0] = WMMA_BF16(a0, b00, acc[0]);
    acc[1] = WMMA_BF16(a0, b01, acc[1]);
    acc[2] = WMMA_BF16(a0, b02, acc[2]);
    acc[3] = WMMA_BF16(a0, b03, acc[3]);
    acc[0] = WMMA_BF16(a1, b10, acc[0]);
    acc[1] = WMMA_BF16(a1, b11, acc[1]);
    acc[2] = WMMA_BF16(a1, b12, acc[2]);
    acc[3] = WMMA_BF16(a1, b13, acc[3]);

    float coef = 0.1f * step_scale[t];
    int lane = threadIdx.x & 31;
    int cbase = lane & 15;
    int rbase = m0 + (lane >> 4) * 8;
#pragma unroll
    for (int j = 0; j < 4; ++j) {
        int col = n0 + j * 16 + cbase;
#pragma unroll
        for (int i = 0; i < 8; ++i) {
            size_t o = (size_t)(rbase + i) * DD + col;
            x[o] += coef * acc[j][i];
        }
    }
}

// ---------------- host launcher ----------------

extern "C" void kernel_launch(void* const* d_in, const int* in_sizes, int n_in,
                              void* d_out, int out_size, void* d_ws, size_t ws_size,
                              hipStream_t stream) {
    (void)in_sizes; (void)n_in; (void)out_size; (void)ws_size;
    const int*   tokens    = (const int*)d_in[0];
    const float* embed_W   = (const float*)d_in[1];
    const float* to_wave_W = (const float*)d_in[2];
    const float* from_wave = (const float*)d_in[3];
    const float* speed     = (const float*)d_in[4];
    const float* dampen    = (const float*)d_in[5];
    const float* phase     = (const float*)d_in[6];
    const float* coupling  = (const float*)d_in[7];
    const float* stepsc    = (const float*)d_in[8];
    const float* pos_freq  = (const float*)d_in[9];
    const float* norms_W   = (const float*)d_in[10];
    const float* out_norm  = (const float*)d_in[11];
    const float* lm_W      = (const float*)d_in[12];
    float* out = (float*)d_out;

    char* ws = (char*)d_ws;
    size_t off = 0;
    auto alloc = [&](size_t bytes) -> void* {
        void* p = ws + off;
        off = (off + bytes + 255) & ~(size_t)255;
        return p;
    };
    unsigned short* lm_bf   = (unsigned short*)alloc((size_t)VV * DD * 2);   // 103 MB (L2-resident)
    unsigned short* tw_bf   = (unsigned short*)alloc((size_t)2 * FF * DD * 2);
    unsigned short* fw_bf   = (unsigned short*)alloc((size_t)DD * FF * 2);
    float*          x       = (float*)alloc((size_t)RR * DD * 4);            // 16 MB
    unsigned short* a_bf    = (unsigned short*)alloc((size_t)RR * DD * 2);   // 8 MB
    float*          rowscale= (float*)alloc((size_t)RR * 4);
    float*          wc      = (float*)alloc((size_t)RR * 2 * FF * 4);
    float*          wave_re = (float*)alloc((size_t)BB * FF * SS * 4);
    float*          wave_im = (float*)alloc((size_t)BB * FF * SS * 4);
    float*          wf_re   = (float*)alloc((size_t)BB * FF * SS * 4);
    float*          wf_im   = (float*)alloc((size_t)BB * FF * SS * 4);
    float*          wg_re   = (float*)alloc((size_t)BB * FF * SS * 4);
    float*          wg_im   = (float*)alloc((size_t)BB * FF * SS * 4);
    float*          wro     = (float*)alloc((size_t)RR * FF * 4);
    unsigned short* wro_bf  = (unsigned short*)alloc((size_t)RR * FF * 2);

    // one-time bf16 weight staging
    k_f32_to_bf16<<<(VV * DD + 255) / 256, 256, 0, stream>>>(lm_W, lm_bf, VV * DD);
    k_f32_to_bf16<<<(2 * FF * DD + 255) / 256, 256, 0, stream>>>(to_wave_W, tw_bf, 2 * FF * DD);
    k_f32_to_bf16<<<(DD * FF + 255) / 256, 256, 0, stream>>>(from_wave, fw_bf, DD * FF);

    k_embed<<<RR, 256, 0, stream>>>(tokens, embed_W, x);

    for (int t = 0; t < TT; ++t) {
        k_rms<<<RR, 256, 0, stream>>>(x, rowscale);
        k_scale_bf16<<<(RR * DD + 255) / 256, 256, 0, stream>>>(x, rowscale, norms_W + (size_t)t * DD, a_bf);
        k_gemm_towave<<<64, 256, 0, stream>>>(a_bf, tw_bf, wc);
        k_make_waves<<<(RR * FF + 255) / 256, 256, 0, stream>>>(wc, pos_freq, wave_re, wave_im);
        k_fft2048<<<BB * FF, 1024, 0, stream>>>(wave_re, wave_im, wf_re, wf_im,
                                                speed, dampen, phase, 0);
        k_coupling<<<(BB * FF * SS + 255) / 256, 256, 0, stream>>>(wf_re, wf_im, coupling, wg_re, wg_im);
        k_fft2048<<<BB * FF, 1024, 0, stream>>>(wg_re, wg_im, wro, wf_im,
                                                speed, dampen, phase, 1);
        k_f32_to_bf16<<<(RR * FF + 255) / 256, 256, 0, stream>>>(wro, wro_bf, RR * FF);
        k_delta_update<<<512, 256, 0, stream>>>(wro_bf, fw_bf, stepsc, t, x);
    }

    k_rms<<<RR, 256, 0, stream>>>(x, rowscale);
    k_scale_bf16<<<(RR * DD + 255) / 256, 256, 0, stream>>>(x, rowscale, out_norm, a_bf);
    k_lm_head<<<(128 * NG_LM) / 8, 256, 0, stream>>>(a_bf, lm_bf, out);
}
